// SplineConv_64707977282144
// MI455X (gfx1250) — matrix operand order repaired and checked by reading, not compiled, not run on hardware
//
#include <hip/hip_runtime.h>
#include <hip/hip_bf16.h>

typedef __attribute__((ext_vector_type(16))) _Float16 v16h;
typedef __attribute__((ext_vector_type(8)))  float    v8f;

#define KS 3
#define NK 9          // KS*KS spline kernels
#define CH 32         // in/out channels

// ---------------------------------------------------------------------------
// f32 -> (hi, lo) f16 split: v ~= hi + lo with ~22 bits of effective mantissa.
// Three WMMA passes (hi*hi + lo*hi + hi*lo) recover ~f32 GEMM precision.
// ---------------------------------------------------------------------------
__device__ __forceinline__ void split_f16(float v, _Float16& hi, _Float16& lo) {
  const _Float16 h = (_Float16)v;
  hi = h;
  lo = (_Float16)(v - (float)h);
}

// ---------------------------------------------------------------------------
// A-fragment loader: 16 nodes x 32 in-channels of x, split into hi/lo f16.
// Layout (16-bit A 16x32, wave32): lane m = lane&15 is row M; per-lane element
// j holds K = 16*(lane>=16) + j  (one contiguous 16-float run).
// ---------------------------------------------------------------------------
__device__ __forceinline__ void load_a_tile(const float* __restrict__ x,
                                            int node_base, int lane,
                                            v16h& a_hi, v16h& a_lo) {
  const int m    = lane & 15;
  const int koff = (lane >> 4) << 3;           // 0 or 8
  const float* xr = x + (size_t)(node_base + m) * CH;
#pragma unroll
  for (int j = 0; j < 16; ++j) {
    const int K = (koff << 1) + j;
    _Float16 hi, lo;
    split_f16(xr[K], hi, lo);
    a_hi[j] = hi;
    a_lo[j] = lo;
  }
}

// D = A*B + C with fp32-emulation (3 chained f16 WMMAs).
__device__ __forceinline__ v8f wmma_f32emu(const v16h& a_hi, const v16h& a_lo,
                                           const v16h& b_hi, const v16h& b_lo,
                                           v8f c) {
  c = __builtin_amdgcn_wmma_f32_16x16x32_f16(false, a_hi, false, b_hi,
                                             (short)0, c, false, false);
  c = __builtin_amdgcn_wmma_f32_16x16x32_f16(false, a_lo, false, b_hi,
                                             (short)0, c, false, false);
  c = __builtin_amdgcn_wmma_f32_16x16x32_f16(false, a_hi, false, b_lo,
                                             (short)0, c, false, false);
  return c;
}

// ---------------------------------------------------------------------------
// Kernel 1: zero the message accumulator (d_out) and degree counters.
// ---------------------------------------------------------------------------
__global__ void spline_zero_kernel(float* __restrict__ out,
                                   float* __restrict__ deg,
                                   int n_out, int n_deg) {
  const int i = blockIdx.x * blockDim.x + threadIdx.x;
  if (i < n_out) out[i] = 0.0f;
  if (i < n_deg) deg[i] = 0.0f;
}

// ---------------------------------------------------------------------------
// Kernel 2: per-node GEMM  y[n,k,:] = x[n,:] @ W_k   (k = 0..8)
// One block = 16 nodes, 9 waves (one per spline kernel k). Each wave does two
// 16x16 output tiles; each tile is 3 chained v_wmma_f32_16x16x32_f16
// (K == IN_CH == 32, fp32-emulated via hi/lo split).
// ---------------------------------------------------------------------------
__global__ void spline_node_gemm_kernel(const float* __restrict__ x,
                                        const float* __restrict__ weight,
                                        float* __restrict__ y) {
  const int lane = threadIdx.x & 31;
  const int k    = threadIdx.x >> 5;           // wave id == spline kernel idx
  const int node_base = blockIdx.x * 16;

  v16h a_hi, a_lo;
  load_a_tile(x, node_base, lane, a_hi, a_lo);

  const float* wk   = weight + k * (CH * CH);  // W_k, [32 in][32 out]
  const int    kb   = (lane >> 4) << 4;        // K base: 0 or 16
  const int    mrow = (lane >> 4) << 3;        // D-row base: 0 or 8

#pragma unroll
  for (int nt = 0; nt < 2; ++nt) {
    const int n = (lane & 15) + nt * 16;       // output channel
    v16h b_hi, b_lo;
#pragma unroll
    for (int j = 0; j < 16; ++j) {             // B 32x16: K = kb+j, N = n
      _Float16 hi, lo;
      split_f16(wk[(kb + j) * CH + n], hi, lo);
      b_hi[j] = hi;
      b_lo[j] = lo;
    }

    v8f c = {};
    c = wmma_f32emu(a_hi, a_lo, b_hi, b_lo, c);

#pragma unroll
    for (int r = 0; r < 8; ++r) {              // D: M = r + mrow, N = n
      const int node = node_base + r + mrow;
      y[(size_t)node * (NK * CH) + k * CH + n] = c[r];
    }
  }
}

// ---------------------------------------------------------------------------
// Kernel 3: per-edge message + scatter. One wave per edge, lane = out channel.
// basis is a degree-1 hat outer product -> <=4 of 9 weights nonzero; the skip
// branches are wave-uniform (all lanes share one edge).
// ---------------------------------------------------------------------------
__global__ void spline_edge_kernel(const float* __restrict__ pseudo,
                                   const int*   __restrict__ ei,
                                   const float* __restrict__ y,
                                   float* __restrict__ out,
                                   float* __restrict__ deg,
                                   int E) {
  const long long t = (long long)blockIdx.x * blockDim.x + threadIdx.x;
  const int e = (int)(t >> 5);
  const int c = (int)(t & 31);
  if (e >= E) return;

  const int row = ei[e];
  const int col = ei[E + e];

  const float p0 = fminf(fmaxf(pseudo[2 * e + 0], 0.0f), 1.0f);
  const float p1 = fminf(fmaxf(pseudo[2 * e + 1], 0.0f), 1.0f);

  float bx[KS], by[KS];
#pragma unroll
  for (int kk = 0; kk < KS; ++kk) {
    const float g = 0.5f * (float)kk;          // knots 0, 0.5, 1
    bx[kk] = fmaxf(1.0f - 2.0f * fabsf(p0 - g), 0.0f);
    by[kk] = fmaxf(1.0f - 2.0f * fabsf(p1 - g), 0.0f);
  }

  const float* yr = y + (size_t)row * (NK * CH) + c;
  float msg = 0.0f;
#pragma unroll
  for (int ka = 0; ka < KS; ++ka) {
    if (bx[ka] > 0.0f) {                       // wave-uniform
#pragma unroll
      for (int kb = 0; kb < KS; ++kb) {
        const float w = bx[ka] * by[kb];
        if (w > 0.0f)                          // wave-uniform
          msg += w * yr[(ka * KS + kb) * CH];
      }
    }
  }

  atomicAdd(&out[(size_t)col * CH + c], msg);
  if (c == 0) atomicAdd(&deg[col], 1.0f);
}

// ---------------------------------------------------------------------------
// Kernel 4: finalize  out = out/max(deg,1) + x @ root_w^T + root_b
// One block = 16 nodes, 2 waves (one per 16-channel output tile); the root
// GEMM is 3 chained WMMAs (fp32-emulated). root_w is [out][in] so each lane's
// 16 B-elements are contiguous in memory.
// ---------------------------------------------------------------------------
__global__ void spline_finalize_kernel(const float* __restrict__ x,
                                       const float* __restrict__ root_w,
                                       const float* __restrict__ root_b,
                                       const float* __restrict__ deg,
                                       float* __restrict__ out) {
  const int lane = threadIdx.x & 31;
  const int nt   = threadIdx.x >> 5;           // 0 or 1: output tile
  const int node_base = blockIdx.x * 16;

  v16h a_hi, a_lo;
  load_a_tile(x, node_base, lane, a_hi, a_lo);

  const int n    = (lane & 15) + nt * 16;      // output channel
  const int kb   = (lane >> 4) << 4;
  const int mrow = (lane >> 4) << 3;

  const float* rw = root_w + (size_t)n * CH + kb;   // contiguous K run
  v16h b_hi, b_lo;
#pragma unroll
  for (int j = 0; j < 16; ++j) {
    _Float16 hi, lo;
    split_f16(rw[j], hi, lo);
    b_hi[j] = hi;
    b_lo[j] = lo;
  }

  v8f c = {};
  c = wmma_f32emu(a_hi, a_lo, b_hi, b_lo, c);

  const float bias = root_b[n];
#pragma unroll
  for (int r = 0; r < 8; ++r) {
    const int    node = node_base + r + mrow;
    const size_t idx  = (size_t)node * CH + n;
    const float  d    = deg[node];
    out[idx] = out[idx] / fmaxf(d, 1.0f) + c[r] + bias;
  }
}

// ---------------------------------------------------------------------------
// Host-side launch. Inputs (setup_inputs order):
//   0: x [N,32] f32   1: pseudo [E,2] f32   2: weight [9,32,32] f32
//   3: root_w [32,32] f32   4: root_b [32] f32   5: edge_index [2,E] int32
// Workspace: y [N,9,32] f32 (115.2 MB) + degree [N] f32.
// ---------------------------------------------------------------------------
extern "C" void kernel_launch(void* const* d_in, const int* in_sizes, int n_in,
                              void* d_out, int out_size, void* d_ws, size_t ws_size,
                              hipStream_t stream) {
  const float* x      = (const float*)d_in[0];
  const float* pseudo = (const float*)d_in[1];
  const float* weight = (const float*)d_in[2];
  const float* root_w = (const float*)d_in[3];
  const float* root_b = (const float*)d_in[4];
  const int*   ei     = (const int*)d_in[5];
  float*       out    = (float*)d_out;

  const int N = in_sizes[0] / CH;
  const int E = in_sizes[1] / 2;

  float* y   = (float*)d_ws;                   // N * 9 * 32 floats
  float* deg = y + (size_t)N * (NK * CH);      // N floats

  // 1) zero accumulators
  {
    const int n_out = N * CH;                  // n_out >= N covers deg too
    spline_zero_kernel<<<(n_out + 255) / 256, 256, 0, stream>>>(out, deg, n_out, N);
  }

  // 2) per-node 9-kernel GEMM via fp32-emulated WMMA (N divisible by 16 here)
  {
    const int ntiles = N / 16;
    spline_node_gemm_kernel<<<ntiles, NK * 32, 0, stream>>>(x, weight, y);
  }

  // 3) per-edge gather/scatter (one wave per edge)
  {
    const long long tot = (long long)E * 32;
    const int blocks = (int)((tot + 255) / 256);
    spline_edge_kernel<<<blocks, 256, 0, stream>>>(pseudo, ei, y, out, deg, E);
  }

  // 4) mean-normalize + root linear via fp32-emulated WMMA
  {
    const int ntiles = N / 16;
    spline_finalize_kernel<<<ntiles, 64, 0, stream>>>(x, root_w, root_b, deg, out);
  }
}
